// PuzzleSim_34471407518202
// MI455X (gfx1250) — compile-verified, matrix operand order checked by reading, not run on hardware
//
#include <hip/hip_runtime.h>
#include <hip/hip_bf16.h>
#include <math.h>

typedef __attribute__((ext_vector_type(16))) _Float16 v16h;
typedef __attribute__((ext_vector_type(8)))  float    v8f;

#define NREF 32
#define CCH  128
#define HH   48
#define WW   48
#define P    (HH * WW)        // 2304 image positions
#define MT   (P / 16)         // 144 M tiles
#define QTOT (NREF * P)       // 73728 reference positions
#define NT   (QTOT / 16)      // 4608 N tiles
#define KC   4                // K chunks of 32 (K = 128)

// CDNA5 16-bit A/B operand layout for 16x16x32 WMMA (wave32):
// lane L<16 holds row/col r=L, K in {0..7} (elems 0..7) and {16..23} (elems 8..15);
// lane L>=16 holds r=L-16, K in {8..15} and {24..31}.
__device__ __forceinline__ void wmma_slot(int r, int k, int& lane, int& elem) {
    int hi = k >> 4;          // which 16-wide K half
    int lo = k & 15;
    if (lo < 8) { lane = r;      elem = hi * 8 + lo; }
    else        { lane = r + 16; elem = hi * 8 + (lo - 8); }
}

// Monotone float -> uint map so unsigned atomicMax preserves float ordering.
__device__ __forceinline__ unsigned f2ord(float f) {
    unsigned u = __float_as_uint(f);
    return (u & 0x80000000u) ? ~u : (u | 0x80000000u);
}

// ---- Phase 1a: normalize img over C, pack as f16 WMMA-A tiles; init output ----
__global__ void norm_pack_img(const float* __restrict__ img,
                              _Float16* __restrict__ Abuf,
                              unsigned* __restrict__ out_u) {
    int p = blockIdx.x * blockDim.x + threadIdx.x;
    if (p >= P) return;
    float s = 0.f;
    for (int c = 0; c < CCH; ++c) { float v = img[c * P + p]; s += v * v; }
    float inv = 1.0f / fmaxf(sqrtf(s), 1e-12f);
    int tile = p >> 4, r = p & 15;
    for (int c = 0; c < CCH; ++c) {
        int kc = c >> 5, k = c & 31, lane, elem;
        wmma_slot(r, k, lane, elem);
        Abuf[((size_t)(tile * KC + kc) * 32 + lane) * 16 + elem] =
            (_Float16)(img[c * P + p] * inv);
    }
    out_u[p] = f2ord(-__builtin_inff());
}

// ---- Phase 1b: normalize refs over C, pack as f16 WMMA-B tiles ----
__global__ void norm_pack_refs(const float* __restrict__ refs,
                               _Float16* __restrict__ Bbuf) {
    int tid = blockIdx.x * blockDim.x + threadIdx.x;
    if (tid >= QTOT) return;
    int n = tid / P, p = tid % P;
    const float* base = refs + (size_t)n * CCH * P + p;
    float s = 0.f;
    for (int c = 0; c < CCH; ++c) { float v = base[c * P]; s += v * v; }
    float inv = 1.0f / fmaxf(sqrtf(s), 1e-12f);
    int q = n * P + p;                  // column index in the big GEMM
    int tile = q >> 4, r = q & 15;
    for (int c = 0; c < CCH; ++c) {
        int kc = c >> 5, k = c & 31, lane, elem;
        wmma_slot(r, k, lane, elem);
        Bbuf[((size_t)(tile * KC + kc) * 32 + lane) * 16 + elem] =
            (_Float16)(base[c * P] * inv);
    }
}

// ---- Phase 2: WMMA GEMM + running max, combined across blocks via atomicMax ----
__global__ void __launch_bounds__(256)
sim_max(const _Float16* __restrict__ Abuf,
        const _Float16* __restrict__ Bbuf,
        unsigned* __restrict__ out_u) {
    const int lane   = threadIdx.x & 31;
    const int wave   = threadIdx.x >> 5;
    const int mgroup = blockIdx.x >> 5;      // 0..17  (144/8 m-groups)
    const int nslice = blockIdx.x & 31;      // 0..31  (4608/144 slices)
    const int mtile  = mgroup * 8 + wave;    // 0..143
    const int nt0    = nslice * (NT / 32);   // 144 N-tiles per slice

    // A operand for this wave's M tile: K = 128 -> 4 x v16h
    v16h a[KC];
#pragma unroll
    for (int kc = 0; kc < KC; ++kc)
        a[kc] = *(const v16h*)(Abuf + ((size_t)(mtile * KC + kc) * 32 + lane) * 16);

    float rmax[8];
#pragma unroll
    for (int v = 0; v < 8; ++v) rmax[v] = -__builtin_inff();

    for (int nt = nt0; nt < nt0 + NT / 32; ++nt) {
        const _Float16* bp = Bbuf + ((size_t)nt * KC * 32 + lane) * 16;
        v16h b0 = *(const v16h*)(bp);
        v16h b1 = *(const v16h*)(bp + 1 * 32 * 16);
        v16h b2 = *(const v16h*)(bp + 2 * 32 * 16);
        v16h b3 = *(const v16h*)(bp + 3 * 32 * 16);

        v8f c0 = {};
        v8f c1 = {};
        // Two independent accumulation chains for ILP; sum at the end.
        c0 = __builtin_amdgcn_wmma_f32_16x16x32_f16(false, a[0], false, b0,
                                                    (short)0, c0, false, false);
        c1 = __builtin_amdgcn_wmma_f32_16x16x32_f16(false, a[1], false, b1,
                                                    (short)0, c1, false, false);
        c0 = __builtin_amdgcn_wmma_f32_16x16x32_f16(false, a[2], false, b2,
                                                    (short)0, c0, false, false);
        c1 = __builtin_amdgcn_wmma_f32_16x16x32_f16(false, a[3], false, b3,
                                                    (short)0, c1, false, false);
        v8f c = c0 + c1;
#pragma unroll
        for (int v = 0; v < 8; ++v) rmax[v] = fmaxf(rmax[v], c[v]);
    }

    // Max over the 16 N-columns: xor shuffles 8,4,2,1 stay inside each 16-lane half.
#pragma unroll
    for (int off = 8; off >= 1; off >>= 1) {
#pragma unroll
        for (int v = 0; v < 8; ++v)
            rmax[v] = fmaxf(rmax[v], __shfl_xor(rmax[v], off, 32));
    }

    // C/D layout: lanes 0-15 -> rows M=v; lanes 16-31 -> rows M=v+8.
    if ((lane & 15) == 0) {
        int mbase = mtile * 16 + ((lane >> 4) ? 8 : 0);
#pragma unroll
        for (int v = 0; v < 8; ++v)
            atomicMax(out_u + (mbase + v), f2ord(rmax[v]));
    }
}

// ---- Phase 3: ordinal uints -> floats, in place in d_out ----
__global__ void finalize_out(unsigned* __restrict__ out_u,
                             float* __restrict__ out_f) {
    int p = blockIdx.x * blockDim.x + threadIdx.x;
    if (p < P) {
        unsigned u = out_u[p];
        unsigned b = (u & 0x80000000u) ? (u & 0x7fffffffu) : ~u;
        out_f[p] = __uint_as_float(b);
    }
}

extern "C" void kernel_launch(void* const* d_in, const int* in_sizes, int n_in,
                              void* d_out, int out_size, void* d_ws, size_t ws_size,
                              hipStream_t stream) {
    const float* refs = (const float*)d_in[0];  // (32,128,48,48) f32
    const float* img  = (const float*)d_in[1];  // (128,48,48)    f32
    // d_in[2] = stride (blocking only; does not affect the max result)

    _Float16* Abuf = (_Float16*)d_ws;                               // 0.59 MB
    _Float16* Bbuf = (_Float16*)((char*)d_ws +
                                 (size_t)MT * KC * 32 * 16 * 2);    // 18.9 MB
    unsigned* out_u = (unsigned*)d_out;

    norm_pack_img <<<P / 256,    256, 0, stream>>>(img,  Abuf, out_u);
    norm_pack_refs<<<QTOT / 256, 256, 0, stream>>>(refs, Bbuf);
    sim_max       <<<(MT / 8) * 32, 256, 0, stream>>>(Abuf, Bbuf, out_u);
    finalize_out  <<<(P + 255) / 256, 256, 0, stream>>>(out_u, (float*)d_out);
}
